// StructureModule_65335042506790
// MI455X (gfx1250) — matrix operand order repaired and checked by reading, not compiled, not run on hardware
//
#include <hip/hip_runtime.h>
#include <hip/hip_bf16.h>
#include <math.h>

// ---------------- problem constants ----------------
#define NRES  768
#define DCH   384
#define PCH   64
#define NH    12
#define KD    32
#define NPQ   4
#define NPV   8
#define FD    1536          // FACTOR*D
#define NATOM 14
#define NSTEP 8
#define FEATD 1536          // H*K + H*PV*3 + H*PV + H*P = 384+288+96+768

typedef __attribute__((ext_vector_type(16))) _Float16 v16h;
typedef __attribute__((ext_vector_type(8)))  float    v8f;

#define WMMA_F16(A_, B_, C_) \
  __builtin_amdgcn_wmma_f32_16x16x32_f16(false, (A_), false, (B_), (short)0, (C_), false, false)

// Build a v16h fragment from 4 float4's (emits v_cvt_pk_f16_f32 pairs)
__device__ __forceinline__ v16h cvt16(float4 a0, float4 a1, float4 a2, float4 a3) {
  v16h r;
  r[0]=(_Float16)a0.x;  r[1]=(_Float16)a0.y;  r[2]=(_Float16)a0.z;  r[3]=(_Float16)a0.w;
  r[4]=(_Float16)a1.x;  r[5]=(_Float16)a1.y;  r[6]=(_Float16)a1.z;  r[7]=(_Float16)a1.w;
  r[8]=(_Float16)a2.x;  r[9]=(_Float16)a2.y;  r[10]=(_Float16)a2.z; r[11]=(_Float16)a2.w;
  r[12]=(_Float16)a3.x; r[13]=(_Float16)a3.y; r[14]=(_Float16)a3.z; r[15]=(_Float16)a3.w;
  return r;
}

// A-fragment (16x32, this file's contiguous k order, same as B): K runs half*16+{0..15}
__device__ __forceinline__ v16h load_a_f32(const float* __restrict__ row, int k0, int half) {
  const float* p = row + k0 + half * 16;
  float4 a0 = *(const float4*)(p);
  float4 a1 = *(const float4*)(p + 4);
  float4 a2 = *(const float4*)(p + 8);
  float4 a3 = *(const float4*)(p + 12);
  return cvt16(a0, a1, a2, a3);
}
__device__ __forceinline__ v16h load_a_f32_scaled(const float* __restrict__ row, int k0,
                                                  int half, float m) {
  const float* p = row + k0 + half * 16;
  float4 a0 = *(const float4*)(p);
  float4 a1 = *(const float4*)(p + 4);
  float4 a2 = *(const float4*)(p + 8);
  float4 a3 = *(const float4*)(p + 12);
  a0.x*=m;a0.y*=m;a0.z*=m;a0.w*=m; a1.x*=m;a1.y*=m;a1.z*=m;a1.w*=m;
  a2.x*=m;a2.y*=m;a2.z*=m;a2.w*=m; a3.x*=m;a3.y*=m;a3.z*=m;a3.w*=m;
  return cvt16(a0, a1, a2, a3);
}
// B-fragment from f32 when the 16 K-values are contiguous in memory
__device__ __forceinline__ v16h load_b_f32(const float* __restrict__ base, int k0, int half) {
  const float* p = base + k0 + half * 16;
  return cvt16(*(const float4*)(p), *(const float4*)(p + 4),
               *(const float4*)(p + 8), *(const float4*)(p + 12));
}
// B-fragment from pre-packed f16 (one 32-byte load)
__device__ __forceinline__ v16h load_b_f16(const _Float16* __restrict__ base, int k0, int half) {
  return *(const v16h*)(base + k0 + half * 16);
}

__device__ __forceinline__ float wred_sum(float v) {
  #pragma unroll
  for (int o = 16; o > 0; o >>= 1) v += __shfl_xor(v, o, 32);
  return v;
}
__device__ __forceinline__ float wred_max(float v) {
  #pragma unroll
  for (int o = 16; o > 0; o >>= 1) v = fmaxf(v, __shfl_xor(v, o, 32));
  return v;
}

// ============================================================
// Weight pack (once): Bt[col*Kd + k] = f16(B[k*Nd + col]), cols padded to Ndp
// ============================================================
__global__ void __launch_bounds__(256) pack_bt_kernel(
    const float* __restrict__ B, _Float16* __restrict__ Bt, int Kd, int Nd, int Ndp) {
  int idx = blockIdx.x * blockDim.x + threadIdx.x;
  if (idx >= Ndp * Kd) return;
  int col = idx / Kd, k = idx % Kd;
  Bt[idx] = (col < Nd) ? (_Float16)B[(size_t)k * Nd + col] : (_Float16)0.f;
}

// ============================================================
// WMMA GEMM, TN tiles per wave sharing one A fragment per k-step.
// C[M,Nd] = A[M,Kd] @ Bt (+bias) (+res); Bt f16 [Ndp,Kd] padded.
// ============================================================
template <int TN>
__global__ void __launch_bounds__(256) gemm_wmma(
    const float* __restrict__ Am, const _Float16* __restrict__ Bt,
    const float* __restrict__ bias, const float* __restrict__ res,
    float* __restrict__ Cm, int M, int Kd, int Nd) {
  int ngrp = ((Nd + 15) >> 4) / TN;
  int wave = blockIdx.x * (blockDim.x >> 5) + (threadIdx.x >> 5);
  if (wave >= (M >> 4) * ngrp) return;
  int mt = wave / ngrp, ng = wave % ngrp;
  int lid = threadIdx.x & 31, half = lid >> 4, sub = lid & 15;
  const float* arow = Am + (size_t)(mt * 16 + sub) * Kd;
  const _Float16* bcol[TN];
  v8f acc[TN];
  #pragma unroll
  for (int t = 0; t < TN; ++t) {
    bcol[t] = Bt + (size_t)((ng * TN + t) * 16 + sub) * Kd;
    acc[t] = (v8f){};
  }
  for (int k0 = 0; k0 < Kd; k0 += 32) {
    v16h a = load_a_f32(arow, k0, half);
    #pragma unroll
    for (int t = 0; t < TN; ++t)
      acc[t] = WMMA_F16(a, load_b_f16(bcol[t], k0, half), acc[t]);
  }
  #pragma unroll
  for (int t = 0; t < TN; ++t) {
    int ocol = (ng * TN + t) * 16 + sub;
    if (ocol < Nd) {
      #pragma unroll
      for (int r = 0; r < 8; ++r) {
        int orow = mt * 16 + half * 8 + r;
        float v = acc[t][r];
        if (bias) v += bias[ocol];
        if (res)  v += res[(size_t)orow * Nd + ocol];
        Cm[(size_t)orow * Nd + ocol] = v;
      }
    }
  }
}

// ============================================================
// LayerNorm (float4 vectorized): wave per row, C % 4 == 0
// ============================================================
__global__ void __launch_bounds__(256) ln_kernel(
    const float* __restrict__ x, const float* __restrict__ s,
    const float* __restrict__ b, float* __restrict__ out, int rows, int C) {
  int row = blockIdx.x * (blockDim.x >> 5) + (threadIdx.x >> 5);
  if (row >= rows) return;
  int lid = threadIdx.x & 31;
  int C4 = C >> 2;
  const float4* x4 = (const float4*)(x + (size_t)row * C);
  float sum = 0.f;
  for (int c = lid; c < C4; c += 32) { float4 v = x4[c]; sum += v.x + v.y + v.z + v.w; }
  float mu = wred_sum(sum) / (float)C;
  float vs = 0.f;
  for (int c = lid; c < C4; c += 32) {
    float4 v = x4[c];
    float d0 = v.x - mu, d1 = v.y - mu, d2 = v.z - mu, d3 = v.w - mu;
    vs += d0 * d0 + d1 * d1 + d2 * d2 + d3 * d3;
  }
  float rstd = rsqrtf(wred_sum(vs) / (float)C + 1e-5f);
  const float4* s4 = (const float4*)s;
  const float4* b4 = (const float4*)b;
  float4* o4 = (float4*)(out + (size_t)row * C);
  for (int c = lid; c < C4; c += 32) {
    float4 v = x4[c], sv = s4[c], bv = b4[c], o;
    o.x = sv.x * (v.x - mu) * rstd + bv.x;
    o.y = sv.y * (v.y - mu) * rstd + bv.y;
    o.z = sv.z * (v.z - mu) * rstd + bv.z;
    o.w = sv.w * (v.w - mu) * rstd + bv.w;
    o4[c] = o;
  }
}

// ============================================================
// Pair LN statistics (once): wave per (i,j) row of 64
// ============================================================
__global__ void __launch_bounds__(256) pair_stats_kernel(
    const float* __restrict__ pair, float* __restrict__ mu, float* __restrict__ rs) {
  int row = blockIdx.x * (blockDim.x >> 5) + (threadIdx.x >> 5);
  if (row >= NRES * NRES) return;
  int lid = threadIdx.x & 31;
  const float* pr = pair + (size_t)row * PCH;
  float x0 = pr[lid], x1 = pr[lid + 32];
  float m = wred_sum(x0 + x1) * (1.f / PCH);
  float d0 = x0 - m, d1 = x1 - m;
  float v = wred_sum(d0 * d0 + d1 * d1) * (1.f / PCH);
  if (lid == 0) { mu[row] = m; rs[row] = rsqrtf(v + 1e-5f); }
}

// ============================================================
// pfT pack (once): pfT[(i*64+p)*N + j] = f16(LN(pair)[i,j,p])
// ============================================================
__global__ void __launch_bounds__(256) pack_pfT_kernel(
    const float* __restrict__ pair, const float* __restrict__ mu,
    const float* __restrict__ rs, const float* __restrict__ ps,
    const float* __restrict__ pb, _Float16* __restrict__ pfT) {
  int ij = blockIdx.x * blockDim.x + threadIdx.x;
  if (ij >= NRES * NRES) return;
  int i = ij / NRES, j = ij % NRES;
  float m = mu[ij], r = rs[ij];
  const float* pr = pair + (size_t)ij * PCH;
  for (int p = 0; p < PCH; ++p) {
    float pf = (pr[p] - m) * r * ps[p] + pb[p];
    pfT[((size_t)i * PCH + p) * NRES + j] = (_Float16)pf;
  }
}

// ============================================================
// Pair bias (once): bias[h,i,j] = sum_p pf(i,j,p) * w_b[p,h]
// ============================================================
__global__ void __launch_bounds__(256) pair_bias_kernel(
    const float* __restrict__ pair, const float* __restrict__ mu,
    const float* __restrict__ rs, const float* __restrict__ ps,
    const float* __restrict__ pb, const float* __restrict__ w_b,
    float* __restrict__ bias) {
  __shared__ float wsh[PCH * NH];
  for (int idx = threadIdx.x; idx < PCH * NH; idx += blockDim.x) wsh[idx] = w_b[idx];
  __syncthreads();
  int ij = blockIdx.x * blockDim.x + threadIdx.x;
  if (ij >= NRES * NRES) return;
  float m = mu[ij], r = rs[ij];
  const float* pr = pair + (size_t)ij * PCH;
  float acc[NH];
  #pragma unroll
  for (int h = 0; h < NH; ++h) acc[h] = 0.f;
  for (int p = 0; p < PCH; ++p) {
    float pf = (pr[p] - m) * r * ps[p] + pb[p];
    #pragma unroll
    for (int h = 0; h < NH; ++h) acc[h] += pf * wsh[p * NH + h];
  }
  int i = ij / NRES, j = ij % NRES;
  #pragma unroll
  for (int h = 0; h < NH; ++h)
    bias[((size_t)h * NRES + i) * NRES + j] = acc[h];
}

// ============================================================
// Rigid frames from backbone atoms (N, CA, C), optional scale
// ============================================================
__global__ void __launch_bounds__(256) frames_kernel(
    const float* __restrict__ pos, float scale,
    float* __restrict__ Rm, float* __restrict__ tv) {
  int i = blockIdx.x * blockDim.x + threadIdx.x;
  if (i >= NRES) return;
  const float* pp = pos + (size_t)i * NATOM * 3;
  float nx = pp[0] * scale, ny = pp[1] * scale, nz = pp[2] * scale;
  float ax = pp[3] * scale, ay = pp[4] * scale, az = pp[5] * scale;
  float cx = pp[6] * scale, cy = pp[7] * scale, cz = pp[8] * scale;
  float d0 = cx - ax, d1 = cy - ay, d2 = cz - az;
  float inv = rsqrtf(d0 * d0 + d1 * d1 + d2 * d2 + 1e-12f);
  float e10 = d0 * inv, e11 = d1 * inv, e12 = d2 * inv;
  float v0 = nx - ax, v1 = ny - ay, v2 = nz - az;
  float dt = e10 * v0 + e11 * v1 + e12 * v2;
  float u0 = v0 - e10 * dt, u1 = v1 - e11 * dt, u2 = v2 - e12 * dt;
  inv = rsqrtf(u0 * u0 + u1 * u1 + u2 * u2 + 1e-12f);
  float e20 = u0 * inv, e21 = u1 * inv, e22 = u2 * inv;
  float e30 = e11 * e22 - e12 * e21;
  float e31 = e12 * e20 - e10 * e22;
  float e32 = e10 * e21 - e11 * e20;
  float* R = Rm + (size_t)i * 9;        // R[r][c]: columns are e1,e2,e3
  R[0] = e10; R[1] = e20; R[2] = e30;
  R[3] = e11; R[4] = e21; R[5] = e31;
  R[6] = e12; R[7] = e22; R[8] = e32;
  tv[(size_t)i * 3 + 0] = ax; tv[(size_t)i * 3 + 1] = ay; tv[(size_t)i * 3 + 2] = az;
}

// ============================================================
// Local points -> global frame, output stride `ostride` per (i,h)
// with zero fill of [np*3, ostride); optional squared norms
// ============================================================
__global__ void __launch_bounds__(256) to_global_kernel(
    const float* __restrict__ loc, const float* __restrict__ Rm,
    const float* __restrict__ tv, float* __restrict__ glob,
    float* __restrict__ sq, int np, int ostride) {
  int idx = blockIdx.x * blockDim.x + threadIdx.x;
  if (idx >= NRES * NH) return;
  int i = idx / NH, h = idx % NH;
  const float* R = Rm + (size_t)i * 9;
  const float* t = tv + (size_t)i * 3;
  float* gp = glob + ((size_t)i * NH + h) * ostride;
  float ss = 0.f;
  for (int p = 0; p < np; ++p) {
    const float* lp = loc + ((size_t)i * NH + h) * np * 3 + p * 3;
    float g0 = R[0] * lp[0] + R[1] * lp[1] + R[2] * lp[2] + t[0];
    float g1 = R[3] * lp[0] + R[4] * lp[1] + R[5] * lp[2] + t[1];
    float g2 = R[6] * lp[0] + R[7] * lp[1] + R[8] * lp[2] + t[2];
    gp[p * 3 + 0] = g0; gp[p * 3 + 1] = g1; gp[p * 3 + 2] = g2;
    ss += g0 * g0 + g1 * g1 + g2 * g2;
  }
  for (int z = np * 3; z < ostride; ++z) gp[z] = 0.f;
  if (sq) sq[idx] = ss;
}

// ============================================================
// v/vp transpose pack (per step): vT[(h*32+c)*N + j]
// ============================================================
__global__ void __launch_bounds__(256) pack_vT_kernel(
    const float* __restrict__ vmat, const float* __restrict__ vp,
    _Float16* __restrict__ vT, _Float16* __restrict__ vpT) {
  int idx = blockIdx.x * blockDim.x + threadIdx.x;
  if (idx >= NH * 32 * NRES) return;
  int j = idx % NRES;
  int hc = idx / NRES;
  int h = hc >> 5, c = hc & 31;
  vT[idx]  = (_Float16)vmat[(size_t)j * (NH * KD) + h * KD + c];
  vpT[idx] = (_Float16)((c < NPV * 3) ? vp[((size_t)j * NH + h) * (NPV * 3) + c] : 0.f);
}

// ============================================================
// Attention logits: wave per (h, i-tile, j-group of 4 tiles).
// q/qp A-fragments loaded once, reused across 8 WMMAs.
// Branch-free epilogue: bitwise mask, hoisted q2/k2/mask loads,
// per-tile base pointers with r*NRES offsets.
// ============================================================
__global__ void __launch_bounds__(256) attn_logits_wmma(
    const float* __restrict__ q, const float* __restrict__ kmat,
    const float* __restrict__ qp, const float* __restrict__ kp,
    const float* __restrict__ q2, const float* __restrict__ k2,
    const float* __restrict__ bias, const float* __restrict__ gamma,
    const unsigned char* __restrict__ mask, float* __restrict__ att) {
  const int nt = NRES / 16, JG = nt / 4;        // 48 tiles, 12 groups
  int wave = blockIdx.x * (blockDim.x >> 5) + (threadIdx.x >> 5);
  if (wave >= NH * nt * JG) return;
  int h = wave / (nt * JG);
  int rem = wave % (nt * JG);
  int it = rem / JG, jg = rem % JG;
  int lid = threadIdx.x & 31, half = lid >> 4, sub = lid & 15;
  int i = it * 16 + sub;
  v16h aq = load_a_f32(q + (size_t)i * (NH * KD) + h * KD, 0, half);
  v16h ap = load_a_f32(qp + ((size_t)i * NH + h) * 32, 0, half);
  float g = log1pf(expf(gamma[h]));               // softplus
  const float wl   = 0.57735026918962576f;        // sqrt(1/3)
  const float coef = 0.5f * wl * 0.23570226039551584f; // wl*wc/2
  const float iscl = 0.17677669529663687f;        // 1/sqrt(K)
  // hoist: per-output-row data (independent of j-tile)
  int row0 = it * 16 + half * 8;
  float q2r[8]; unsigned mir[8];
  #pragma unroll
  for (int r = 0; r < 8; ++r) {
    q2r[r] = q2[(size_t)(row0 + r) * NH + h];
    mir[r] = (unsigned)mask[row0 + r];
  }
  const float* brow = bias + ((size_t)h * NRES + row0) * NRES;
  float*       orow = att  + ((size_t)h * NRES + row0) * NRES;
  #pragma unroll
  for (int t = 0; t < 4; ++t) {
    int jt = jg * 4 + t;
    int j = jt * 16 + sub;
    v16h bk = load_b_f32(kmat + (size_t)j * (NH * KD) + h * KD, 0, half);
    v16h bp = load_b_f32(kp + ((size_t)j * NH + h) * 32, 0, half);
    v8f accS = {}, accP = {};
    accS = WMMA_F16(aq, bk, accS);
    accP = WMMA_F16(ap, bp, accP);
    float k2v = k2[(size_t)j * NH + h];
    unsigned mj = (unsigned)mask[j];
    const float* bp0 = brow + j;
    float*       op0 = orow + j;
    #pragma unroll
    for (int r = 0; r < 8; ++r) {
      float biasv = bp0[(size_t)r * NRES];
      float sc = accS[r] * iscl;
      float d2 = q2r[r] + k2v - 2.f * accP[r];
      float lg = wl * (sc + biasv) - coef * g * d2;
      op0[(size_t)r * NRES] = ((mir[r] & mj) != 0u) ? lg : -1e9f;
    }
  }
}

// ============================================================
// Row softmax (float4): wave per (h,i) row of 768
// ============================================================
__global__ void __launch_bounds__(256) softmax_kernel(float* __restrict__ att) {
  int row = blockIdx.x * (blockDim.x >> 5) + (threadIdx.x >> 5);
  if (row >= NH * NRES) return;
  int lid = threadIdx.x & 31;
  float4* p4 = (float4*)(att + (size_t)row * NRES);
  const int C4 = NRES / 4;
  float mx = -1e30f;
  for (int c = lid; c < C4; c += 32) {
    float4 v = p4[c];
    mx = fmaxf(mx, fmaxf(fmaxf(v.x, v.y), fmaxf(v.z, v.w)));
  }
  mx = wred_max(mx);
  float s = 0.f;
  for (int c = lid; c < C4; c += 32) {
    float4 v = p4[c];
    v.x = __expf(v.x - mx); v.y = __expf(v.y - mx);
    v.z = __expf(v.z - mx); v.w = __expf(v.w - mx);
    p4[c] = v;
    s += v.x + v.y + v.z + v.w;
  }
  s = wred_sum(s);
  float inv = 1.f / s;
  for (int c = lid; c < C4; c += 32) {
    float4 v = p4[c];
    v.x *= inv; v.y *= inv; v.z *= inv; v.w *= inv;
    p4[c] = v;
  }
}

// ============================================================
// Attention apply: wave per (h, i-tile) computes all 4 col-tiles
// (o_s 32 cols + o_pg 24 cols) sharing one att A-fragment per k-step
// ============================================================
__global__ void __launch_bounds__(256) attn_apply_wmma(
    const float* __restrict__ att, const _Float16* __restrict__ vT,
    const _Float16* __restrict__ vpT, float* __restrict__ feat,
    float* __restrict__ opg) {
  const int nt = NRES / 16;
  int wave = blockIdx.x * (blockDim.x >> 5) + (threadIdx.x >> 5);
  if (wave >= NH * nt) return;
  int h = wave / nt, it = wave % nt;
  int lid = threadIdx.x & 31, half = lid >> 4, sub = lid & 15;
  int i = it * 16 + sub;
  const float* arow = att + ((size_t)h * NRES + i) * NRES;
  const _Float16* bcol[4];
  v8f acc[4];
  #pragma unroll
  for (int t = 0; t < 4; ++t) {
    int c32 = (t & 1) * 16 + sub;
    bcol[t] = ((t < 2) ? vT : vpT) + ((size_t)h * 32 + c32) * NRES;
    acc[t] = (v8f){};
  }
  for (int k0 = 0; k0 < NRES; k0 += 32) {
    v16h a = load_a_f32(arow, k0, half);
    #pragma unroll
    for (int t = 0; t < 4; ++t)
      acc[t] = WMMA_F16(a, load_b_f16(bcol[t], k0, half), acc[t]);
  }
  #pragma unroll
  for (int t = 0; t < 4; ++t) {
    int c32 = (t & 1) * 16 + sub;
    #pragma unroll
    for (int r = 0; r < 8; ++r) {
      int oi = it * 16 + half * 8 + r;
      if (t < 2)
        feat[(size_t)oi * FEATD + h * KD + c32] = acc[t][r];
      else if (c32 < NPV * 3)
        opg[((size_t)oi * NH + h) * (NPV * 3) + c32] = acc[t][r];
    }
  }
}

// ============================================================
// o_pair: wave per residue i computes all 4 p-tiles sharing the
// masked att A-fragment; B = one 32-byte pfT load per tile
// ============================================================
__global__ void __launch_bounds__(256) opair_wmma(
    const float* __restrict__ att, const _Float16* __restrict__ pfT,
    float* __restrict__ feat) {
  int wave = blockIdx.x * (blockDim.x >> 5) + (threadIdx.x >> 5);
  if (wave >= NRES) return;
  int i = wave;
  int lid = threadIdx.x & 31, half = lid >> 4, sub = lid & 15;
  int hr = (sub < NH) ? sub : (NH - 1);
  float mval = (sub < NH) ? 1.f : 0.f;
  const float* arow = att + ((size_t)hr * NRES + i) * NRES;
  const _Float16* bcol[4];
  v8f acc[4];
  #pragma unroll
  for (int t = 0; t < 4; ++t) {
    bcol[t] = pfT + ((size_t)i * PCH + t * 16 + sub) * NRES;
    acc[t] = (v8f){};
  }
  for (int k0 = 0; k0 < NRES; k0 += 32) {
    v16h a = load_a_f32_scaled(arow, k0, half, mval);
    #pragma unroll
    for (int t = 0; t < 4; ++t)
      acc[t] = WMMA_F16(a, load_b_f16(bcol[t], k0, half), acc[t]);
  }
  #pragma unroll
  for (int t = 0; t < 4; ++t) {
    int pcol = t * 16 + sub;
    #pragma unroll
    for (int r = 0; r < 8; ++r) {
      int hrow = half * 8 + r;
      if (hrow < NH)
        feat[(size_t)i * FEATD + (NH * KD + NH * NPV * 4) + hrow * PCH + pcol] = acc[t][r];
    }
  }
}

// ============================================================
// o_pl = R^T (o_pg - t) into feat[384..671]; norms into [672..767]
// ============================================================
__global__ void __launch_bounds__(256) finalize_pts_kernel(
    const float* __restrict__ opg, const float* __restrict__ Rm,
    const float* __restrict__ tv, float* __restrict__ feat) {
  int idx = blockIdx.x * blockDim.x + threadIdx.x;
  if (idx >= NRES * NH) return;
  int i = idx / NH, h = idx % NH;
  const float* R = Rm + (size_t)i * 9;
  const float* t = tv + (size_t)i * 3;
  float* f = feat + (size_t)i * FEATD;
  for (int p = 0; p < NPV; ++p) {
    const float* g = opg + ((size_t)i * NH + h) * (NPV * 3) + p * 3;
    float x0 = g[0] - t[0], x1 = g[1] - t[1], x2 = g[2] - t[2];
    float l0 = R[0] * x0 + R[3] * x1 + R[6] * x2;   // R^T x
    float l1 = R[1] * x0 + R[4] * x1 + R[7] * x2;
    float l2 = R[2] * x0 + R[5] * x1 + R[8] * x2;
    int base = NH * KD + (h * NPV + p) * 3;
    f[base + 0] = l0; f[base + 1] = l1; f[base + 2] = l2;
    f[NH * KD + NH * NPV * 3 + h * NPV + p] = sqrtf(l0 * l0 + l1 * l1 + l2 * l2 + 1e-8f);
  }
}

// ============================================================
// data *= gelu(gate)  (tanh approx), float4 vectorized
// ============================================================
__global__ void __launch_bounds__(256) gelu_mul_kernel(
    const float* __restrict__ gate, float* __restrict__ data, int n4) {
  int idx = blockIdx.x * blockDim.x + threadIdx.x;
  if (idx >= n4) return;
  float4 x = ((const float4*)gate)[idx];
  float4 d = ((float4*)data)[idx];
  const float c0 = 0.7978845608028654f, c1 = 0.044715f;
  d.x *= 0.5f * x.x * (1.f + tanhf(c0 * (x.x + c1 * x.x * x.x * x.x)));
  d.y *= 0.5f * x.y * (1.f + tanhf(c0 * (x.y + c1 * x.y * x.y * x.y)));
  d.z *= 0.5f * x.z * (1.f + tanhf(c0 * (x.z + c1 * x.z * x.z * x.z)));
  d.w *= 0.5f * x.w * (1.f + tanhf(c0 * (x.w + c1 * x.w * x.w * x.w)));
  ((float4*)data)[idx] = d;
}

// ============================================================
// pos update: new = R @ (10*upd) + t ; also write traj slice
// ============================================================
__global__ void __launch_bounds__(256) update_pos_kernel(
    const float* __restrict__ upd, const float* __restrict__ Rm,
    const float* __restrict__ tv, float* __restrict__ pos,
    float* __restrict__ traj) {
  int idx = blockIdx.x * blockDim.x + threadIdx.x;
  if (idx >= NRES * NATOM) return;
  int i = idx / NATOM;
  const float* u = upd + (size_t)idx * 3;
  const float* R = Rm + (size_t)i * 9;
  const float* t = tv + (size_t)i * 3;
  float x0 = 10.f * u[0], x1 = 10.f * u[1], x2 = 10.f * u[2];
  float g0 = R[0] * x0 + R[1] * x1 + R[2] * x2 + t[0];
  float g1 = R[3] * x0 + R[4] * x1 + R[5] * x2 + t[1];
  float g2 = R[6] * x0 + R[7] * x1 + R[8] * x2 + t[2];
  float* pp = pos + (size_t)idx * 3;
  float* tj = traj + (size_t)idx * 3;
  pp[0] = g0; pp[1] = g1; pp[2] = g2;
  tj[0] = g0; tj[1] = g1; tj[2] = g2;
}

// ============================================================
// Host side
// ============================================================
static inline void launch_gemm(const float* A, const _Float16* Bt, const float* bias,
                               const float* res, float* C, int M, int Kd, int Nd,
                               hipStream_t st) {
  int ntile = (Nd + 15) / 16;
  if (ntile % 4 == 0) {
    int waves = (M / 16) * (ntile / 4);
    gemm_wmma<4><<<(waves + 7) / 8, 256, 0, st>>>(A, Bt, bias, res, C, M, Kd, Nd);
  } else if (ntile % 3 == 0) {
    int waves = (M / 16) * (ntile / 3);
    gemm_wmma<3><<<(waves + 7) / 8, 256, 0, st>>>(A, Bt, bias, res, C, M, Kd, Nd);
  } else {
    int waves = (M / 16) * ntile;
    gemm_wmma<1><<<(waves + 7) / 8, 256, 0, st>>>(A, Bt, bias, res, C, M, Kd, Nd);
  }
}
static inline void launch_pack(const float* B, _Float16* Bt, int Kd, int Nd, int Ndp,
                               hipStream_t st) {
  pack_bt_kernel<<<(Ndp * Kd + 255) / 256, 256, 0, st>>>(B, Bt, Kd, Nd, Ndp);
}

extern "C" void kernel_launch(void* const* d_in, const int* in_sizes, int n_in,
                              void* d_out, int out_size, void* d_ws, size_t ws_size,
                              hipStream_t stream) {
  const float* in_local = (const float*)d_in[0];
  const float* in_pair  = (const float*)d_in[1];
  const float* in_pos   = (const float*)d_in[2];
  const unsigned char* in_mask = (const unsigned char*)d_in[3];
  int pi = 4;
  const float* p_lnl_s = (const float*)d_in[pi++];
  const float* p_lnl_b = (const float*)d_in[pi++];
  const float* p_lnp_s = (const float*)d_in[pi++];
  const float* p_lnp_b = (const float*)d_in[pi++];
  const float* w_q   = (const float*)d_in[pi++];
  const float* w_k   = (const float*)d_in[pi++];
  const float* w_v   = (const float*)d_in[pi++];
  const float* w_qp  = (const float*)d_in[pi++];
  const float* w_kp  = (const float*)d_in[pi++];
  const float* w_vp  = (const float*)d_in[pi++];
  const float* w_b   = (const float*)d_in[pi++];
  const float* gamma = (const float*)d_in[pi++];
  const float* w_o   = (const float*)d_in[pi++];
  const float* b_o   = (const float*)d_in[pi++];
  const float* ln2_s = (const float*)d_in[pi++];
  const float* ln2_b = (const float*)d_in[pi++];
  const float* w_gate = (const float*)d_in[pi++];
  const float* w_data = (const float*)d_in[pi++];
  const float* w_uo   = (const float*)d_in[pi++];
  const float* ln3_s  = (const float*)d_in[pi++];
  const float* ln3_b  = (const float*)d_in[pi++];
  const float* w_pos  = (const float*)d_in[pi++];
  const float* lno_s  = (const float*)d_in[pi++];
  const float* lno_b  = (const float*)d_in[pi++];

  float* ws = (float*)d_ws;
  size_t off = 0;
  auto alloc = [&](size_t nfloats) {
    float* p = ws + off; off += (nfloats + 7) & ~(size_t)7; return p;
  };
  auto allocH = [&](size_t nhalfs) {
    float* p = ws + off; off += ((nhalfs + 1) / 2 + 7) & ~(size_t)7;
    return (_Float16*)p;
  };
  float* local = alloc((size_t)NRES * DCH);
  float* lf    = alloc((size_t)NRES * DCH);
  float* Rm    = alloc((size_t)NRES * 9);
  float* tv    = alloc((size_t)NRES * 3);
  float* R2    = alloc((size_t)NRES * 9);
  float* t2    = alloc((size_t)NRES * 3);
  float* q     = alloc((size_t)NRES * NH * KD);
  float* kmat  = alloc((size_t)NRES * NH * KD);
  float* vmat  = alloc((size_t)NRES * NH * KD);
  float* ptsl  = alloc((size_t)NRES * NH * NPV * 3);
  float* qp    = alloc((size_t)NRES * NH * 32);   // padded to 32 per (i,h)
  float* kp    = alloc((size_t)NRES * NH * 32);
  float* vp    = alloc((size_t)NRES * NH * NPV * 3);
  float* q2    = alloc((size_t)NRES * NH);
  float* k2    = alloc((size_t)NRES * NH);
  float* mu    = alloc((size_t)NRES * NRES);
  float* rs    = alloc((size_t)NRES * NRES);
  float* bias  = alloc((size_t)NH * NRES * NRES);
  float* att   = alloc((size_t)NH * NRES * NRES);
  float* feat  = alloc((size_t)NRES * FEATD);
  float* opg   = alloc((size_t)NRES * NH * NPV * 3);
  float* tmp1  = alloc((size_t)NRES * DCH);
  float* gate  = alloc((size_t)NRES * FD);
  float* dat   = alloc((size_t)NRES * FD);
  float* posb  = alloc((size_t)NRES * NATOM * 3);
  float* updb  = alloc((size_t)NRES * NATOM * 3);
  // f16 packs
  _Float16* bt_q    = allocH((size_t)(NH * KD) * DCH);
  _Float16* bt_k    = allocH((size_t)(NH * KD) * DCH);
  _Float16* bt_v    = allocH((size_t)(NH * KD) * DCH);
  _Float16* bt_qp   = allocH((size_t)(NH * NPQ * 3) * DCH);
  _Float16* bt_kp   = allocH((size_t)(NH * NPQ * 3) * DCH);
  _Float16* bt_vp   = allocH((size_t)(NH * NPV * 3) * DCH);
  _Float16* bt_o    = allocH((size_t)DCH * FEATD);
  _Float16* bt_gate = allocH((size_t)FD * DCH);
  _Float16* bt_data = allocH((size_t)FD * DCH);
  _Float16* bt_uo   = allocH((size_t)DCH * FD);
  _Float16* bt_pos  = allocH((size_t)48 * DCH);        // 42 padded to 48
  _Float16* vT      = allocH((size_t)NH * 32 * NRES);
  _Float16* vpT     = allocH((size_t)NH * 32 * NRES);
  _Float16* pfT     = allocH((size_t)NRES * PCH * NRES);
  (void)ws_size; (void)in_sizes; (void)n_in;

  float* out_ln   = (float*)d_out;                               // [N,D]
  float* out_pos  = out_ln + (size_t)NRES * DCH;                 // [N,A,3]
  float* out_traj = out_pos + (size_t)NRES * NATOM * 3;          // [S,N,A,3]

  hipMemcpyAsync(local, in_local, sizeof(float) * NRES * DCH,
                 hipMemcpyDeviceToDevice, stream);
  hipMemcpyAsync(posb, in_pos, sizeof(float) * NRES * NATOM * 3,
                 hipMemcpyDeviceToDevice, stream);

  // -------- loop-invariant precompute --------
  launch_pack(w_q,    bt_q,    DCH, NH * KD,      NH * KD,      stream);
  launch_pack(w_k,    bt_k,    DCH, NH * KD,      NH * KD,      stream);
  launch_pack(w_v,    bt_v,    DCH, NH * KD,      NH * KD,      stream);
  launch_pack(w_qp,   bt_qp,   DCH, NH * NPQ * 3, NH * NPQ * 3, stream);
  launch_pack(w_kp,   bt_kp,   DCH, NH * NPQ * 3, NH * NPQ * 3, stream);
  launch_pack(w_vp,   bt_vp,   DCH, NH * NPV * 3, NH * NPV * 3, stream);
  launch_pack(w_o,    bt_o,    FEATD, DCH, DCH, stream);
  launch_pack(w_gate, bt_gate, DCH, FD, FD, stream);
  launch_pack(w_data, bt_data, DCH, FD, FD, stream);
  launch_pack(w_uo,   bt_uo,   FD, DCH, DCH, stream);
  launch_pack(w_pos,  bt_pos,  DCH, NATOM * 3, 48, stream);

  pair_stats_kernel<<<(NRES * NRES + 7) / 8, 256, 0, stream>>>(in_pair, mu, rs);
  pack_pfT_kernel<<<(NRES * NRES + 255) / 256, 256, 0, stream>>>(
      in_pair, mu, rs, p_lnp_s, p_lnp_b, pfT);
  pair_bias_kernel<<<(NRES * NRES + 255) / 256, 256, 0, stream>>>(
      in_pair, mu, rs, p_lnp_s, p_lnp_b, w_b, bias);

  for (int s = 0; s < NSTEP; ++s) {
    frames_kernel<<<(NRES + 255) / 256, 256, 0, stream>>>(posb, 0.1f, Rm, tv);
    ln_kernel<<<(NRES + 7) / 8, 256, 0, stream>>>(local, p_lnl_s, p_lnl_b, lf, NRES, DCH);

    launch_gemm(lf, bt_q, nullptr, nullptr, q, NRES, DCH, NH * KD, stream);
    launch_gemm(lf, bt_k, nullptr, nullptr, kmat, NRES, DCH, NH * KD, stream);
    launch_gemm(lf, bt_v, nullptr, nullptr, vmat, NRES, DCH, NH * KD, stream);

    launch_gemm(lf, bt_qp, nullptr, nullptr, ptsl, NRES, DCH, NH * NPQ * 3, stream);
    to_global_kernel<<<(NRES * NH + 255) / 256, 256, 0, stream>>>(ptsl, Rm, tv, qp, q2, NPQ, 32);
    launch_gemm(lf, bt_kp, nullptr, nullptr, ptsl, NRES, DCH, NH * NPQ * 3, stream);
    to_global_kernel<<<(NRES * NH + 255) / 256, 256, 0, stream>>>(ptsl, Rm, tv, kp, k2, NPQ, 32);
    launch_gemm(lf, bt_vp, nullptr, nullptr, ptsl, NRES, DCH, NH * NPV * 3, stream);
    to_global_kernel<<<(NRES * NH + 255) / 256, 256, 0, stream>>>(ptsl, Rm, tv, vp, nullptr,
                                                                  NPV, NPV * 3);
    pack_vT_kernel<<<(NH * 32 * NRES + 255) / 256, 256, 0, stream>>>(vmat, vp, vT, vpT);

    {
      int waves = NH * (NRES / 16) * (NRES / 64);
      attn_logits_wmma<<<(waves + 7) / 8, 256, 0, stream>>>(
          q, kmat, qp, kp, q2, k2, bias, gamma, in_mask, att);
    }
    softmax_kernel<<<(NH * NRES + 7) / 8, 256, 0, stream>>>(att);
    {
      int waves = NH * (NRES / 16);
      attn_apply_wmma<<<(waves + 7) / 8, 256, 0, stream>>>(att, vT, vpT, feat, opg);
    }
    opair_wmma<<<(NRES + 7) / 8, 256, 0, stream>>>(att, pfT, feat);
    finalize_pts_kernel<<<(NRES * NH + 255) / 256, 256, 0, stream>>>(opg, Rm, tv, feat);

    // local += feat @ w_o + b_o   (in-place residual)
    launch_gemm(feat, bt_o, b_o, local, local, NRES, FEATD, DCH, stream);

    // transition MLP: local += (data * gelu(gate)) @ w_uo
    ln_kernel<<<(NRES + 7) / 8, 256, 0, stream>>>(local, ln2_s, ln2_b, tmp1, NRES, DCH);
    launch_gemm(tmp1, bt_gate, nullptr, nullptr, gate, NRES, DCH, FD, stream);
    launch_gemm(tmp1, bt_data, nullptr, nullptr, dat, NRES, DCH, FD, stream);
    gelu_mul_kernel<<<(NRES * FD / 4 + 255) / 256, 256, 0, stream>>>(gate, dat, NRES * FD / 4);
    launch_gemm(dat, bt_uo, nullptr, local, local, NRES, FD, DCH, stream);

    // position update (frames at scale 1.0)
    ln_kernel<<<(NRES + 7) / 8, 256, 0, stream>>>(local, ln3_s, ln3_b, tmp1, NRES, DCH);
    launch_gemm(tmp1, bt_pos, nullptr, nullptr, updb, NRES, DCH, NATOM * 3, stream);
    frames_kernel<<<(NRES + 255) / 256, 256, 0, stream>>>(posb, 1.0f, R2, t2);
    update_pos_kernel<<<(NRES * NATOM + 255) / 256, 256, 0, stream>>>(
        updb, R2, t2, posb, out_traj + (size_t)s * NRES * NATOM * 3);
  }

  ln_kernel<<<(NRES + 7) / 8, 256, 0, stream>>>(local, lno_s, lno_b, out_ln, NRES, DCH);
  hipMemcpyAsync(out_pos, posb, sizeof(float) * NRES * NATOM * 3,
                 hipMemcpyDeviceToDevice, stream);
  (void)out_size;
}
